// LSTM_53334903882078
// MI455X (gfx1250) — matrix operand order repaired and checked by reading, not compiled, not run on hardware
//
#include <hip/hip_runtime.h>

#define HID 512
#define BATCH 1024
#define PRED 48

typedef __attribute__((ext_vector_type(16))) __bf16 v16bf;
typedef __attribute__((ext_vector_type(8)))  __bf16 v8bf;
typedef __attribute__((ext_vector_type(8)))  float  v8f;

__device__ __forceinline__ v16bf load2x8(const __bf16* p0, const __bf16* p1) {
  v8bf lo = *(const v8bf*)p0;
  v8bf hi = *(const v8bf*)p1;
  v16bf r;
#pragma unroll
  for (int i = 0; i < 8; ++i) { r[i] = lo[i]; r[i + 8] = hi[i]; }
  return r;
}

__device__ __forceinline__ float fsigmoid(float x) {
  return 1.0f / (1.0f + __expf(-x));
}
__device__ __forceinline__ float ftanh(float x) {
  return 1.0f - 2.0f / (__expf(2.0f * x) + 1.0f);
}

// ---------------- init / prep kernels ----------------

__global__ void init_state_kernel(float* c, float* h_f32, __bf16* h_bf0, __bf16* h_bf1, int n) {
  int i = blockIdx.x * blockDim.x + threadIdx.x;
  if (i < n) {
    c[i] = 0.0f;
    h_f32[i] = 0.0f;
    h_bf0[i] = (__bf16)0.0f;
    h_bf1[i] = (__bf16)0.0f;
  }
}

__global__ void prep_weights_kernel(const float* __restrict__ w, __bf16* __restrict__ wbf, int n) {
  int i = blockIdx.x * blockDim.x + threadIdx.x;
  if (i < n) wbf[i] = (__bf16)w[i];
}

__global__ void prep_bias_kernel(const float* __restrict__ b_ih, const float* __restrict__ b_hh,
                                 float* __restrict__ bias, int n) {
  int i = blockIdx.x * blockDim.x + threadIdx.x;
  if (i < n) bias[i] = b_ih[i] + b_hh[i];
}

// ---------------- fused LSTM cell (one timestep) ----------------
// grid = dim3(8, 32), block = 128 (4 waves).
// wave: cols [jbase, jbase+16) for all 4 gates, rows [rowBase, rowBase+32).
// K loop: software-pipelined ping-pong (2 chunks in flight), chunk 0 peeled
// with literal-zero C operand so accumulators need no init movs.

// Load one K-chunk (32 wide): 2 A tiles + 4 gate B tiles.
#define LOADCH(KK, AB, BB)                                          \
  do {                                                              \
    const int k0_ = (KK) + half * 16;                               \
    AB[0] = load2x8(arow0 + k0_, arow0 + k0_ + 8);                  \
    AB[1] = load2x8(arow1 + k0_, arow1 + k0_ + 8);                  \
    const int kb_ = (KK) + half * 16;                               \
    BB[0] = *(const v16bf*)(wrowg[0] + kb_);                        \
    BB[1] = *(const v16bf*)(wrowg[1] + kb_);                        \
    BB[2] = *(const v16bf*)(wrowg[2] + kb_);                        \
    BB[3] = *(const v16bf*)(wrowg[3] + kb_);                        \
  } while (0)

// 8 accumulate WMMAs for one chunk (4 gates x 2 M-tiles).
#define WMMACH(AB, BB)                                                        \
  do {                                                                        \
    _Pragma("unroll")                                                         \
    for (int g_ = 0; g_ < 4; ++g_) {                                          \
      acc[g_][0] = __builtin_amdgcn_wmma_f32_16x16x32_bf16(                   \
          false, AB[0], false, BB[g_], (short)0, acc[g_][0], false, false);   \
      acc[g_][1] = __builtin_amdgcn_wmma_f32_16x16x32_bf16(                   \
          false, AB[1], false, BB[g_], (short)0, acc[g_][1], false, false);   \
    }                                                                         \
  } while (0)

__global__ void __launch_bounds__(128, 1)
lstm_cell_kernel(const __bf16* __restrict__ h_in,   // [B, H] bf16
                 const __bf16* __restrict__ w_bf,   // [4H, H] bf16 (w_hh)
                 const float*  __restrict__ xt,     // per-batch input
                 int xt_stride,
                 const float*  __restrict__ w_ih,   // [4H] (IN==1)
                 const float*  __restrict__ bias,   // [4H] = b_ih + b_hh
                 float*        __restrict__ c_st,   // [B, H] in-place
                 float*        __restrict__ h_f32,  // [B, H] out
                 __bf16*       __restrict__ h_out)  // [B, H] out (next step A)
{
  const int lane    = threadIdx.x & 31;
  const int wave    = threadIdx.x >> 5;
  const int ntile   = blockIdx.x * 4 + wave;   // 0..31
  const int jbase   = ntile * 16;
  const int rowBase = blockIdx.y * 32;         // 0..992 step 32
  const int half    = lane >> 4;               // 0 | 1
  const int lrow    = lane & 15;

  // A-matrix rows (contiguous K order, same as B: lanes 0-15 K {0..15},
  // lanes 16-31 K {16..31}).
  const __bf16* arow0 = h_in + (size_t)(rowBase + lrow) * HID;
  const __bf16* arow1 = arow0 + (size_t)16 * HID;

  // B column n (=lrow) of gate g is w_hh row (g*H + jbase + n); K-contiguous.
  const __bf16* wrowg[4];
#pragma unroll
  for (int g = 0; g < 4; ++g)
    wrowg[g] = w_bf + (size_t)(g * HID + jbase + lrow) * HID;

  v8f acc[4][2];
  v16bf A0[2], B0[4];   // ping
  v16bf A1[2], B1[4];   // pong

  // prologue: chunks 0 and 1 in flight
  LOADCH(0, A0, B0);
  LOADCH(32, A1, B1);

  // chunk 0: C operand is literal zero (no accumulator init movs)
  {
    const v8f zed = {};
#pragma unroll
    for (int g = 0; g < 4; ++g) {
      acc[g][0] = __builtin_amdgcn_wmma_f32_16x16x32_bf16(
          false, A0[0], false, B0[g], (short)0, zed, false, false);
      acc[g][1] = __builtin_amdgcn_wmma_f32_16x16x32_bf16(
          false, A0[1], false, B0[g], (short)0, zed, false, false);
    }
  }

  // steady state: chunks 1..12 in pairs; two chunks always in flight.
#pragma unroll 1
  for (int it = 1; it < (HID / 32) - 4; it += 2) {   // it = 1,3,...,11
    const int kk = it * 32;
    LOADCH(kk + 32, A0, B0);   // chunk it+1
    WMMACH(A1, B1);            // chunk it
    LOADCH(kk + 64, A1, B1);   // chunk it+2
    WMMACH(A0, B0);            // chunk it+1
  }

  // tail: chunks 13, 14, 15 (pong currently holds chunk 13)
  LOADCH(HID - 64, A0, B0);    // chunk 14
  WMMACH(A1, B1);              // chunk 13
  LOADCH(HID - 32, A1, B1);    // chunk 15
  WMMACH(A0, B0);              // chunk 14
  WMMACH(A1, B1);              // chunk 15

  // Epilogue: C/D layout — lane half selects rows {0..7} vs {8..15}, col = lrow.
  const int j = jbase + lrow;
  float wi[4], bi[4];
#pragma unroll
  for (int g = 0; g < 4; ++g) {
    wi[g] = w_ih[g * HID + j];
    bi[g] = bias[g * HID + j];
  }

#pragma unroll
  for (int m = 0; m < 2; ++m) {
#pragma unroll
    for (int r = 0; r < 8; ++r) {
      const int row = rowBase + m * 16 + half * 8 + r;
      const float xv = xt[(size_t)row * xt_stride];
      const float gi = acc[0][m][r] + xv * wi[0] + bi[0];
      const float gf = acc[1][m][r] + xv * wi[1] + bi[1];
      const float gg = acc[2][m][r] + xv * wi[2] + bi[2];
      const float go = acc[3][m][r] + xv * wi[3] + bi[3];
      const size_t idx = (size_t)row * HID + j;
      const float cn = fsigmoid(gf) * c_st[idx] + fsigmoid(gi) * ftanh(gg);
      const float hn = fsigmoid(go) * ftanh(cn);
      c_st[idx]  = cn;
      h_f32[idx] = hn;
      h_out[idx] = (__bf16)hn;
    }
  }
}

// ---------------- decoder fc: out = h @ w_fc.T + b_fc (OUT==1) ----------------
// one wave per batch row; block=128 -> 4 rows/block, grid=256.
__global__ void fc_kernel(const float* __restrict__ h_f32,
                          const float* __restrict__ w_fc,
                          const float* __restrict__ b_fc,
                          float* __restrict__ xt_buf,
                          float* __restrict__ out, int step) {
  const int row  = blockIdx.x * (blockDim.x >> 5) + (threadIdx.x >> 5);
  const int lane = threadIdx.x & 31;
  float s = 0.0f;
  const float* hr = h_f32 + (size_t)row * HID;
  for (int k = lane; k < HID; k += 32) s += hr[k] * w_fc[k];
#pragma unroll
  for (int off = 16; off > 0; off >>= 1) s += __shfl_down(s, off, 32);
  if (lane == 0) {
    const float v = s + b_fc[0];
    xt_buf[row] = v;
    out[(size_t)row * PRED + step] = v;
  }
}

// ---------------- host launch ----------------

extern "C" void kernel_launch(void* const* d_in, const int* in_sizes, int n_in,
                              void* d_out, int out_size, void* d_ws, size_t ws_size,
                              hipStream_t stream) {
  const float* x    = (const float*)d_in[0];  // [1024, 512, 1]
  const float* w_ih = (const float*)d_in[1];  // [2048, 1]
  const float* w_hh = (const float*)d_in[2];  // [2048, 512]
  const float* b_ih = (const float*)d_in[3];  // [2048]
  const float* b_hh = (const float*)d_in[4];  // [2048]
  const float* w_fc = (const float*)d_in[5];  // [1, 512]
  const float* b_fc = (const float*)d_in[6];  // [1]
  float* out = (float*)d_out;                 // [1024, 48]

  char* ws = (char*)d_ws;
  float*  c_st   = (float*) (ws);                           // 2 MB
  float*  h_f32  = (float*) (ws + (2u << 20));              // 2 MB
  __bf16* h_bf0  = (__bf16*)(ws + (4u << 20));              // 1 MB
  __bf16* h_bf1  = (__bf16*)(ws + (5u << 20));              // 1 MB
  __bf16* w_bf   = (__bf16*)(ws + (6u << 20));              // 2 MB
  float*  bias   = (float*) (ws + (8u << 20));              // 8 KB
  float*  xt_buf = (float*) (ws + (8u << 20) + (16u << 10));// 4 KB

  // --- one-time prep (inside capture; deterministic) ---
  {
    const int n = BATCH * HID;
    init_state_kernel<<<(n + 255) / 256, 256, 0, stream>>>(c_st, h_f32, h_bf0, h_bf1, n);
    const int nw = 4 * HID * HID;
    prep_weights_kernel<<<(nw + 255) / 256, 256, 0, stream>>>(w_hh, w_bf, nw);
    prep_bias_kernel<<<(4 * HID + 255) / 256, 256, 0, stream>>>(b_ih, b_hh, bias, 4 * HID);
  }

  const dim3 cell_grid(8, 32);
  __bf16* hin  = h_bf0;
  __bf16* hout = h_bf1;

  // --- encoder: 512 steps, xt[b] = x[b*512 + t] ---
  for (int t = 0; t < 512; ++t) {
    lstm_cell_kernel<<<cell_grid, 128, 0, stream>>>(
        hin, w_bf, x + t, 512, w_ih, bias, c_st, h_f32, hout);
    __bf16* tmp = hin; hin = hout; hout = tmp;
  }

  // --- autoregressive decoder: fc(h) -> out & feedback, then cell ---
  for (int p = 0; p < PRED; ++p) {
    fc_kernel<<<BATCH / 4, 128, 0, stream>>>(h_f32, w_fc, b_fc, xt_buf, out, p);
    lstm_cell_kernel<<<cell_grid, 128, 0, stream>>>(
        hin, w_bf, xt_buf, 1, w_ih, bias, c_st, h_f32, hout);
    __bf16* tmp = hin; hin = hout; hout = tmp;
  }
}
